// HierarchicalTreeLSTMs_36086315221306
// MI455X (gfx1250) — compile-verified, matrix operand order checked by reading, not run on hardware
//
#include <hip/hip_runtime.h>

#define H 256

typedef __attribute__((ext_vector_type(16))) __bf16 v16bf;
typedef __attribute__((ext_vector_type(8)))  float  v8f;

union ABf { v16bf v; uint4 q[2]; };

__device__ __forceinline__ unsigned short f2bf(float f) {
  unsigned u = __float_as_uint(f);
  u += 0x7FFFu + ((u >> 16) & 1u);
  return (unsigned short)(u >> 16);
}
__device__ __forceinline__ float fsig(float x)  { return 1.f / (1.f + __expf(-x)); }
__device__ __forceinline__ float ftanh(float x) { return 1.f - 2.f / (__expf(2.f * x) + 1.f); }

__device__ __forceinline__ v8f wmma_bf(const ABf& a, const ABf& b, v8f c) {
  return __builtin_amdgcn_wmma_f32_16x16x32_bf16(false, a.v, false, b.v, (short)0, c, false, false);
}

// ---------------- small helpers ----------------
__global__ void k_cvt_bf16(const float* __restrict__ in, unsigned short* __restrict__ out, int n) {
  int i = blockIdx.x * blockDim.x + threadIdx.x;
  if (i < n) out[i] = f2bf(in[i]);
}
__global__ void k_bias_sum(const float* __restrict__ a, const float* __restrict__ b,
                           float* __restrict__ o, int n) {
  int i = blockIdx.x * blockDim.x + threadIdx.x;
  if (i < n) o[i] = a[i] + b[i];
}

// ---------------- 32-row x 4-gate GEMM slab, low B-register footprint ----------------
// Wave w owns hidden cols [32w, 32w+32): gate g's columns are g*256 + hc.
// Per gate: one B pair (16 VGPRs) reused by 4 WMMAs.
__device__ __forceinline__ void gemm_slab4(const unsigned short* s_a,
                                           const unsigned short* __restrict__ W,
                                           int hb, int mlane, int kb,
                                           v8f (&acc)[2][4][2]) {
#pragma unroll
  for (int kk = 0; kk < 8; ++kk) {
    ABf A0, A1;
    const unsigned short* a0 = s_a + mlane * H + kk * 32 + kb;
    const unsigned short* a1 = a0 + 16 * H;
    A0.q[0] = *(const uint4*)a0;  A0.q[1] = *(const uint4*)(a0 + 16);
    A1.q[0] = *(const uint4*)a1;  A1.q[1] = *(const uint4*)(a1 + 16);
#pragma unroll
    for (int g = 0; g < 4; ++g) {
      ABf B0, B1;
      const unsigned short* bp0 = W + (size_t)(g * 256 + hb + mlane) * H + kk * 32 + kb;
      const unsigned short* bp1 = bp0 + 16 * H;
      B0.q[0] = *(const uint4*)bp0;  B0.q[1] = *(const uint4*)(bp0 + 16);
      B1.q[0] = *(const uint4*)bp1;  B1.q[1] = *(const uint4*)(bp1 + 16);
      acc[0][g][0] = wmma_bf(A0, B0, acc[0][g][0]);
      acc[1][g][0] = wmma_bf(A1, B0, acc[1][g][0]);
      acc[0][g][1] = wmma_bf(A0, B1, acc[0][g][1]);
      acc[1][g][1] = wmma_bf(A1, B1, acc[1][g][1]);
    }
  }
}

// ---------------- encoder for a 32-node LDS tile ----------------
// s_elr: [32][512] bf16 = concat(el, er). Wenc row-major [256][512] bf16.
__device__ __forceinline__ void encode32(const unsigned short* s_elr,
                                         const unsigned short* __restrict__ Wenc,
                                         const float* __restrict__ benc,
                                         unsigned short* __restrict__ ctx,
                                         int off, int i0, int n, float* out0) {
  const int tid   = threadIdx.x;
  const int lane  = tid & 31;
  const int w     = tid >> 5;
  const int mlane = lane & 15;
  const int kb    = (lane >> 4) ? 8 : 0;
  const int mbase = (lane >> 4) ? 8 : 0;

  v8f acc[2][2];
#pragma unroll
  for (int mt = 0; mt < 2; ++mt)
#pragma unroll
    for (int tt = 0; tt < 2; ++tt) acc[mt][tt] = (v8f){0.f,0.f,0.f,0.f,0.f,0.f,0.f,0.f};

#pragma unroll
  for (int kk = 0; kk < 16; ++kk) {
    ABf A0, A1;
    const unsigned short* a0 = s_elr + mlane * (2 * H) + kk * 32 + kb;
    const unsigned short* a1 = a0 + 16 * (2 * H);
    A0.q[0] = *(const uint4*)a0;  A0.q[1] = *(const uint4*)(a0 + 16);
    A1.q[0] = *(const uint4*)a1;  A1.q[1] = *(const uint4*)(a1 + 16);
    ABf B0, B1;
    const unsigned short* bp0 = Wenc + (size_t)(w * 32 + mlane) * (2 * H) + kk * 32 + kb;
    const unsigned short* bp1 = bp0 + 16 * (2 * H);
    B0.q[0] = *(const uint4*)bp0;  B0.q[1] = *(const uint4*)(bp0 + 16);
    B1.q[0] = *(const uint4*)bp1;  B1.q[1] = *(const uint4*)(bp1 + 16);
    acc[0][0] = wmma_bf(A0, B0, acc[0][0]);
    acc[1][0] = wmma_bf(A1, B0, acc[1][0]);
    acc[0][1] = wmma_bf(A0, B1, acc[0][1]);
    acc[1][1] = wmma_bf(A1, B1, acc[1][1]);
  }

  if (i0 + 32 <= n) {
    // fast path: no guards, constant-offset stores
#pragma unroll
    for (int tt = 0; tt < 2; ++tt) {
      int coln = w * 32 + tt * 16 + mlane;
      float bv = benc[coln];
      unsigned short* cp = ctx + (size_t)(off + i0 + mbase) * H + coln;
#pragma unroll
      for (int mt = 0; mt < 2; ++mt)
#pragma unroll
        for (int r = 0; r < 8; ++r) {
          float v = ftanh(acc[mt][tt][r] + bv);
          cp[(mt * 16 + r) * H] = f2bf(v);
        }
    }
  } else {
    // partial tile (top of tree); also handles the root f32 output
#pragma unroll
    for (int tt = 0; tt < 2; ++tt) {
      int coln = w * 32 + tt * 16 + mlane;
      float bv = benc[coln];
#pragma unroll
      for (int mt = 0; mt < 2; ++mt)
#pragma unroll
        for (int r = 0; r < 8; ++r) {
          int i = i0 + mt * 16 + mbase + r;
          if (i < n) {
            float v = ftanh(acc[mt][tt][r] + bv);
            ctx[(size_t)(off + i) * H + coln] = f2bf(v);
            if (out0 && i == 0) out0[coln] = v;
          }
        }
    }
  }
}

// ---------------- leaf level: fused single-step LSTMs (i,g,o only; f*c0=0) + encoder ----------------
__global__ __launch_bounds__(256)
void k_leaf(unsigned short* __restrict__ ctx,
            const unsigned short* __restrict__ WihL, const unsigned short* __restrict__ WihR,
            const float* __restrict__ bsumL, const float* __restrict__ bsumR,
            const unsigned short* __restrict__ Wenc, const float* __restrict__ benc,
            int off, int n) {
  __shared__ unsigned short s_x[32 * H];        // 16 KB
  __shared__ unsigned short s_elr[32 * 2 * H];  // 32 KB

  const int tid = threadIdx.x;
  const int i0  = blockIdx.x * 32;

  for (int q = tid; q < 32 * 32; q += 256) {
    int m = q >> 5, ch = q & 31;
    *(uint4*)&s_x[m * H + ch * 8] = *(const uint4*)&ctx[(size_t)(off + i0 + m) * H + ch * 8];
  }
  __syncthreads();

  const int lane  = tid & 31;
  const int w     = tid >> 5;
  const int mlane = lane & 15;
  const int kb    = (lane >> 4) ? 8 : 0;
  const int mbase = (lane >> 4) ? 8 : 0;
  const int hb    = w * 32;
  const int gbase[3] = {0, 512, 768};           // i, g, o column bases

  for (int side = 0; side < 2; ++side) {
    const unsigned short* Wih = side ? WihR : WihL;
    const float* bsum = side ? bsumR : bsumL;

    v8f acc[2][3][2];
#pragma unroll
    for (int mt = 0; mt < 2; ++mt)
#pragma unroll
      for (int g = 0; g < 3; ++g)
#pragma unroll
        for (int tt = 0; tt < 2; ++tt) acc[mt][g][tt] = (v8f){0.f,0.f,0.f,0.f,0.f,0.f,0.f,0.f};

#pragma unroll
    for (int kk = 0; kk < 8; ++kk) {
      ABf A0, A1;
      const unsigned short* a0 = s_x + mlane * H + kk * 32 + kb;
      const unsigned short* a1 = a0 + 16 * H;
      A0.q[0] = *(const uint4*)a0;  A0.q[1] = *(const uint4*)(a0 + 16);
      A1.q[0] = *(const uint4*)a1;  A1.q[1] = *(const uint4*)(a1 + 16);
#pragma unroll
      for (int g = 0; g < 3; ++g) {
        ABf B0, B1;
        const unsigned short* bp0 = Wih + (size_t)(gbase[g] + hb + mlane) * H + kk * 32 + kb;
        const unsigned short* bp1 = bp0 + 16 * H;
        B0.q[0] = *(const uint4*)bp0;  B0.q[1] = *(const uint4*)(bp0 + 16);
        B1.q[0] = *(const uint4*)bp1;  B1.q[1] = *(const uint4*)(bp1 + 16);
        acc[0][g][0] = wmma_bf(A0, B0, acc[0][g][0]);
        acc[1][g][0] = wmma_bf(A1, B0, acc[1][g][0]);
        acc[0][g][1] = wmma_bf(A0, B1, acc[0][g][1]);
        acc[1][g][1] = wmma_bf(A1, B1, acc[1][g][1]);
      }
    }

#pragma unroll
    for (int tt = 0; tt < 2; ++tt) {
      int hc = hb + tt * 16 + mlane;
      float bvi = bsum[hc], bvg = bsum[512 + hc], bvo = bsum[768 + hc];
      unsigned short* eb = s_elr + mbase * (2 * H) + side * H + hc;
#pragma unroll
      for (int mt = 0; mt < 2; ++mt)
#pragma unroll
        for (int r = 0; r < 8; ++r) {
          float cn = fsig(acc[mt][0][tt][r] + bvi) * ftanh(acc[mt][1][tt][r] + bvg);
          float hn = fsig(acc[mt][2][tt][r] + bvo) * ftanh(cn);
          eb[(mt * 16 + r) * (2 * H)] = f2bf(hn);
        }
    }
  }
  __syncthreads();
  encode32(s_elr, Wenc, benc, ctx, off, i0, n, nullptr);
}

// ---------------- inner levels: whole level fused (5 timesteps x 2 sides + encoder) ----------------
// h and c never leave the WGP: s_h (bf16) and s_c (f32) live in LDS; c elements are wave-private.
__global__ __launch_bounds__(256)
void k_level(unsigned short* __restrict__ ctx,
             const unsigned short* __restrict__ WihL, const unsigned short* __restrict__ WhhL,
             const unsigned short* __restrict__ WihR, const unsigned short* __restrict__ WhhR,
             const float* __restrict__ bsumL, const float* __restrict__ bsumR,
             const unsigned short* __restrict__ Wenc, const float* __restrict__ benc,
             int off, int coff, int n, float* out0) {
  __shared__ unsigned short s_x[32 * H];        // 16 KB
  __shared__ unsigned short s_h[32 * H];        // 16 KB
  __shared__ float          s_c[32 * H];        // 32 KB
  __shared__ unsigned short s_elr[32 * 2 * H];  // 32 KB

  const int tid   = threadIdx.x;
  const int i0    = blockIdx.x * 32;
  const int lane  = tid & 31;
  const int w     = tid >> 5;
  const int mlane = lane & 15;
  const int kb    = (lane >> 4) ? 8 : 0;
  const int mbase = (lane >> 4) ? 8 : 0;
  const int hb    = w * 32;

  for (int side = 0; side < 2; ++side) {
    const unsigned short* Wih  = side ? WihR : WihL;
    const unsigned short* Whh  = side ? WhhR : WhhL;
    const float*          bsum = side ? bsumR : bsumL;

    float bv[4][2];
#pragma unroll
    for (int g = 0; g < 4; ++g)
#pragma unroll
      for (int tt = 0; tt < 2; ++tt) bv[g][tt] = bsum[g * 256 + hb + tt * 16 + mlane];

    for (int t = 0; t < 5; ++t) {
      __syncthreads();   // prior phase done with s_x
      // stage x tile (L chain: c3,c2,c1,c0,own ; R chain: own,c4,c5,c6,c7)
      for (int q = tid; q < 32 * 32; q += 256) {
        int m = q >> 5, ch = q & 31;
        int i = i0 + m;
        int xrow;
        if (side == 0) xrow = (t < 4) ? (coff + i * 8 + (3 - t)) : (off + i);
        else           xrow = (t == 0) ? (off + i) : (coff + i * 8 + (t + 3));
        *(uint4*)&s_x[m * H + ch * 8] = *(const uint4*)&ctx[(size_t)xrow * H + ch * 8];
      }
      __syncthreads();

      v8f acc[2][4][2];
#pragma unroll
      for (int mt = 0; mt < 2; ++mt)
#pragma unroll
        for (int g = 0; g < 4; ++g)
#pragma unroll
          for (int tt = 0; tt < 2; ++tt) acc[mt][g][tt] = (v8f){0.f,0.f,0.f,0.f,0.f,0.f,0.f,0.f};

      gemm_slab4(s_x, Wih, hb, mlane, kb, acc);
      if (t > 0) gemm_slab4(s_h, Whh, hb, mlane, kb, acc);
      __syncthreads();   // all waves done reading s_h before rewrite

      // in-register cell update; c is wave-private in s_c
#pragma unroll
      for (int tt = 0; tt < 2; ++tt) {
        int hc = hb + tt * 16 + mlane;
#pragma unroll
        for (int mt = 0; mt < 2; ++mt)
#pragma unroll
          for (int r = 0; r < 8; ++r) {
            int M = mt * 16 + mbase + r;
            float ig = acc[mt][0][tt][r] + bv[0][tt];
            float fg = acc[mt][1][tt][r] + bv[1][tt];
            float gg = acc[mt][2][tt][r] + bv[2][tt];
            float og = acc[mt][3][tt][r] + bv[3][tt];
            float cold = (t == 0) ? 0.f : s_c[M * H + hc];
            float cn = fsig(fg) * cold + fsig(ig) * ftanh(gg);
            float hn = fsig(og) * ftanh(cn);
            if (t < 4) {
              s_c[M * H + hc] = cn;
              s_h[M * H + hc] = f2bf(hn);
            } else {
              s_elr[M * (2 * H) + side * H + hc] = f2bf(hn);
            }
          }
      }
    }
  }
  __syncthreads();
  encode32(s_elr, Wenc, benc, ctx, off, i0, n, out0);
}

// ---------------- host orchestration ----------------
extern "C" void kernel_launch(void* const* d_in, const int* in_sizes, int n_in,
                              void* d_out, int out_size, void* d_ws, size_t ws_size,
                              hipStream_t stream) {
  (void)in_sizes; (void)n_in; (void)out_size; (void)ws_size;
  const float* node_init = (const float*)d_in[0];
  const float* fWih_l = (const float*)d_in[1];
  const float* fWhh_l = (const float*)d_in[2];
  const float* fbih_l = (const float*)d_in[3];
  const float* fbhh_l = (const float*)d_in[4];
  const float* fWih_r = (const float*)d_in[5];
  const float* fWhh_r = (const float*)d_in[6];
  const float* fbih_r = (const float*)d_in[7];
  const float* fbhh_r = (const float*)d_in[8];
  const float* fWenc  = (const float*)d_in[9];
  const float* fbenc  = (const float*)d_in[10];

  const int NT = 299593;
  static const int OFF[7] = {0, 1, 9, 73, 585, 4681, 37449};
  static const int SZ[7]  = {1, 8, 64, 512, 4096, 32768, 262144};

  size_t p = (size_t)d_ws;
  auto carve = [&](size_t bytes) -> void* {
    size_t r = p;
    p += (bytes + 255) & ~(size_t)255;
    return (void*)r;
  };
  unsigned short* ctx  = (unsigned short*)carve((size_t)NT * 256 * 2);
  unsigned short* wihl = (unsigned short*)carve((size_t)1024 * 256 * 2);
  unsigned short* whhl = (unsigned short*)carve((size_t)1024 * 256 * 2);
  unsigned short* wihr = (unsigned short*)carve((size_t)1024 * 256 * 2);
  unsigned short* whhr = (unsigned short*)carve((size_t)1024 * 256 * 2);
  unsigned short* wenc = (unsigned short*)carve((size_t)256 * 512 * 2);
  float* bsl = (float*)carve(1024 * 4);
  float* bsr = (float*)carve(1024 * 4);

  auto cvt = [&](const float* src, unsigned short* dst, int n) {
    k_cvt_bf16<<<(n + 255) / 256, 256, 0, stream>>>(src, dst, n);
  };
  cvt(node_init, ctx, NT * 256);
  cvt(fWih_l, wihl, 1024 * 256);
  cvt(fWhh_l, whhl, 1024 * 256);
  cvt(fWih_r, wihr, 1024 * 256);
  cvt(fWhh_r, whhr, 1024 * 256);
  cvt(fWenc,  wenc, 256 * 512);
  k_bias_sum<<<4, 256, 0, stream>>>(fbih_l, fbhh_l, bsl, 1024);
  k_bias_sum<<<4, 256, 0, stream>>>(fbih_r, fbhh_r, bsr, 1024);

  // Leaf level d=6: fused single-step LSTMs + encoder (f-gate skipped since c0=0)
  k_leaf<<<dim3(SZ[6] / 32), 256, 0, stream>>>(ctx, wihl, wihr, bsl, bsr, wenc, fbenc,
                                               OFF[6], SZ[6]);

  // Inner levels d=5..0: one fully fused kernel per level
  for (int d = 5; d >= 0; --d) {
    int n = SZ[d], off = OFF[d], coff = OFF[d + 1];
    int tiles = (n + 31) / 32;
    k_level<<<dim3(tiles), 256, 0, stream>>>(ctx, wihl, whhl, wihr, whhr, bsl, bsr,
                                             wenc, fbenc, off, coff, n,
                                             (d == 0) ? (float*)d_out : nullptr);
  }
}
